// BinarizedLinear_41360535061123
// MI455X (gfx1250) — matrix operand order repaired and checked by reading, not compiled, not run on hardware
//
#include <hip/hip_runtime.h>
#include <hip/hip_bf16.h>

typedef __attribute__((ext_vector_type(16))) _Float16 v16h;
typedef __attribute__((ext_vector_type(8)))  float    v8f;
typedef __attribute__((ext_vector_type(4)))  float    v4f;

#define D_DIRS  128
#define OUT_F   1024
#define IN_F    1024
#define XWORDS  (IN_F / 32)          // 32 words per direction
#define NOISE_SCALE 0.1f

// ---------------------------------------------------------------------------
// Phase 1: s[o,i] = sigmoid(weight[o,i])  (4 MB, L2-resident, reused 128x)
// ---------------------------------------------------------------------------
__global__ void sigmoid_kernel(const float* __restrict__ w,
                               float* __restrict__ s, int n) {
    int i = blockIdx.x * blockDim.x + threadIdx.x;
    if (i < n) s[i] = 1.0f / (1.0f + __expf(-w[i]));
}

// ---------------------------------------------------------------------------
// Phase 2: bit-pack x[d,i] (int 0/1) into xw[d][i/32]
// ---------------------------------------------------------------------------
__global__ void pack_x_kernel(const int* __restrict__ x,
                              unsigned* __restrict__ xw, int nwords) {
    int wi = blockIdx.x * blockDim.x + threadIdx.x;
    if (wi < nwords) {
        const int* p = x + (size_t)wi * 32;
        unsigned bits = 0u;
#pragma unroll
        for (int b = 0; b < 32; ++b)
            bits |= (p[b] != 0 ? 1u : 0u) << b;
        xw[wi] = bits;
    }
}

// ---------------------------------------------------------------------------
// Phase 3: streaming Bernoulli-GEMV via WMMA.
//   A[16x32] f16 = (u_w < s)          (pure compare -> select)
//   B[32x16] f16 = x[d, i0+k] broadcast across N  (lane L <-> N=L%16)
//   C += A x B  -> every column of C holds  sum_k (u<s)*x[k]
// One wave per (direction d, 16-output tile), K-chunks of 32.
// ---------------------------------------------------------------------------
__global__ __launch_bounds__(256) void bin_forward_kernel(
    const float*    __restrict__ u_w,   // [D, OUT, IN]  streaming, NT
    const float*    __restrict__ bias,  // [OUT]
    const float*    __restrict__ u_b,   // [D, OUT]
    const float*    __restrict__ s,     // [OUT, IN]  (workspace, L2-resident)
    const unsigned* __restrict__ xw,    // [D, IN/32] (workspace)
    float*          __restrict__ out)   // [D, OUT]
{
    const int wave = blockIdx.x * (blockDim.x >> 5) + (threadIdx.x >> 5);
    const int lane = threadIdx.x & 31;

    const int d  = wave >> 6;          // 64 tiles of 16 outputs per direction
    const int o0 = (wave & 63) << 4;
    const int m  = lane & 15;          // A-matrix row this lane supplies
    const int r  = o0 + m;             // absolute output row
    // 16-bit A-matrix 16x32 layout: lanes 0-15 cover K={0..7,16..23},
    // lanes 16-31 cover K={8..15,24..31}.
    const int k0 = (lane & 16) ? 8 : 0;

    const float*    pu = u_w + ((size_t)d * OUT_F + (size_t)r) * IN_F;
    const float*    ps = s   + (size_t)r * IN_F;
    const unsigned* px = xw  + (size_t)d * XWORDS;

    const _Float16 one_h  = (_Float16)1.0f;
    const _Float16 zero_h = (_Float16)0.0f;

    v8f acc = {};

    for (int i0 = 0; i0 < IN_F; i0 += 32) {
        // prefetch the u_w stream ahead (speculative, NT path)
        __builtin_prefetch(pu + i0 + 256, 0, 0);

        v4f u0 = __builtin_nontemporal_load((const v4f*)(pu + i0 + k0));
        v4f u1 = __builtin_nontemporal_load((const v4f*)(pu + i0 + k0 + 4));
        v4f u2 = __builtin_nontemporal_load((const v4f*)(pu + i0 + k0 + 16));
        v4f u3 = __builtin_nontemporal_load((const v4f*)(pu + i0 + k0 + 20));
        v4f t0 = *(const v4f*)(ps + i0 + k0);
        v4f t1 = *(const v4f*)(ps + i0 + k0 + 4);
        v4f t2 = *(const v4f*)(ps + i0 + k0 + 16);
        v4f t3 = *(const v4f*)(ps + i0 + k0 + 20);
        const unsigned xb = px[i0 >> 5];

        // A = Bernoulli bits (no x masking): cmp + select only
        v16h a;
#pragma unroll
        for (int e = 0; e < 4; ++e) {
            a[e]      = (u0[e] < t0[e]) ? one_h : zero_h;  // K = k0 + e
            a[4 + e]  = (u1[e] < t1[e]) ? one_h : zero_h;  // K = k0 + 4 + e
            a[8 + e]  = (u2[e] < t2[e]) ? one_h : zero_h;  // K = k0 + 16 + e
            a[12 + e] = (u3[e] < t3[e]) ? one_h : zero_h;  // K = k0 + 20 + e
        }

        // B[k][n] = x bit k, broadcast over all 16 columns.
        // 32x16 f16 B layout: lane L holds N=L%16, halves s cover K = k0+s (s<8), k0+8+s (s>=8).
        unsigned bwords[8];
#pragma unroll
        for (int j = 0; j < 8; ++j) {
            const int kk = k0 + 2 * j + ((j & 4) ? 8 : 0);
            bwords[j] = (((xb >> kk) & 1u) ? 0x3C00u : 0u) |
                        (((xb >> (kk + 1)) & 1u) ? 0x3C000000u : 0u);
        }
        v16h bmat = *(const v16h*)bwords;

        acc = __builtin_amdgcn_wmma_f32_16x16x32_f16(
            /*neg_a=*/false, a, /*neg_b=*/false, bmat,
            /*c_mod=*/(short)0, acc, /*reuse_a=*/false, /*reuse_b=*/false);
    }

    // C/D layout: VGPR j holds M=j (lanes 0-15) and M=j+8 (lanes 16-31);
    // all 16 columns are identical, so lane 0 / lane 16 each emit 8 results.
    if (lane == 0 || lane == 16) {
        const int mbase = (lane == 0) ? 0 : 8;
#pragma unroll
        for (int j = 0; j < 8; ++j) {
            const int    o  = o0 + mbase + j;
            const size_t oi = (size_t)d * OUT_F + o;
            const float  th = bias[o] + (u_b[oi] - 0.5f) * NOISE_SCALE;
            out[oi] = (acc[j] > th) ? 1.0f : 0.0f;
        }
    }
}

// ---------------------------------------------------------------------------
extern "C" void kernel_launch(void* const* d_in, const int* in_sizes, int n_in,
                              void* d_out, int out_size, void* d_ws, size_t ws_size,
                              hipStream_t stream) {
    const int*   x      = (const int*)  d_in[0];  // [D, IN] bool -> int
    const float* weight = (const float*)d_in[1];  // [OUT, IN]
    const float* bias   = (const float*)d_in[2];  // [OUT]
    const float* u_w    = (const float*)d_in[3];  // [D, OUT, IN]
    const float* u_b    = (const float*)d_in[4];  // [D, OUT]
    float*       out    = (float*)d_out;          // [D, OUT]

    float*    s  = (float*)d_ws;                                      // 4 MB
    unsigned* xw = (unsigned*)((char*)d_ws +
                               (size_t)OUT_F * IN_F * sizeof(float)); // 16 KB

    // Phase 1: sigmoid(weight)
    {
        const int n = OUT_F * IN_F;
        sigmoid_kernel<<<(n + 255) / 256, 256, 0, stream>>>(weight, s, n);
    }
    // Phase 2: pack x bits
    {
        const int nwords = D_DIRS * XWORDS;
        pack_x_kernel<<<(nwords + 255) / 256, 256, 0, stream>>>(x, xw, nwords);
    }
    // Phase 3: 8192 waves (1 per (d, 16-output tile)), 8 waves per block
    {
        const int tiles  = D_DIRS * (OUT_F / 16);   // 8192
        const int blocks = tiles / 8;               // 1024
        bin_forward_kernel<<<blocks, 256, 0, stream>>>(u_w, bias, u_b, s, xw, out);
    }
}
